// SoftTokenMixer_21337397527127
// MI455X (gfx1250) — matrix operand order repaired and checked by reading, not compiled, not run on hardware
//
#include <hip/hip_runtime.h>
#include <stdint.h>

#define VOCAB   50257
#define EMBED   512
#define ROWS    4096                         // BATCH * SEQ
#define KCH     32                           // WMMA K (bf16)
#define NC64    ((VOCAB + 63) / 64)          // 786 K=64 iterations
#define PADC    (2 * NC64)                   // 1572 padded K=32 chunks in the B slab
#define MROWS   32                           // rows per workgroup (2 M-tiles)
#define ASTRIDE 20                           // dwords per staged-A row: 16B-aligned, conflict-free

typedef __attribute__((ext_vector_type(16))) __bf16 v16bf;
typedef __attribute__((ext_vector_type(8)))  float  v8f;
typedef __attribute__((ext_vector_type(4), aligned(4))) float f4u; // dword-aligned b128 load

union FragBF { uint4 u4[2]; v16bf v; };

__device__ __forceinline__ uint32_t pack_bf16(float a, float b) {
    uint32_t x = __builtin_bit_cast(uint32_t, a);
    uint32_t y = __builtin_bit_cast(uint32_t, b);
    x = (x + 0x7FFFu + ((x >> 16) & 1u)) >> 16;   // round-to-nearest-even
    y = (y + 0x7FFFu + ((y >> 16) & 1u)) >> 16;
    return (x & 0xFFFFu) | (y << 16);
}

// Workgroup barrier that only drains the DS counter (A-tile visibility).
// Unlike __syncthreads() it leaves global loads (B frags / logits) in flight
// across the barrier; their s_wait_loadcnt lands at first use (the WMMAs).
__device__ __forceinline__ void lds_barrier() {
    asm volatile("s_wait_dscnt 0x0\n\t"
                 "s_barrier_signal -1\n\t"
                 "s_barrier_wait -1" ::: "memory");
}

// ---------------------------------------------------------------------------
// One-time W (f32 row-major) -> bf16 slab pre-swizzled into per-lane
// V_WMMA_*_BF16 B fragments in contiguous k order (k = 16*(lane/16) + elem); padded to PADC K=32 chunks.
// Slab = 51.5 MB -> resident in the 192 MB L2.
// ---------------------------------------------------------------------------
__global__ void prep_wswizzle(const float* __restrict__ W, uint32_t* __restrict__ Bw) {
    __shared__ float lw[32 * 68];
    const int tid = threadIdx.x;
    const int kc  = blockIdx.x >> 3;        // K=32 chunk 0..PADC-1
    const int cb  = blockIdx.x & 7;         // 64-column block
    const int c0  = cb * 64;
    {
        const int r  = tid >> 3;            // K-row 0..31
        const int cq = (tid & 7) * 8;
        const int v  = kc * KCH + r;
        #pragma unroll
        for (int i = 0; i < 8; ++i) {
            float x = 0.0f;
            if (v < VOCAB) x = W[(size_t)v * EMBED + c0 + cq + i];
            lw[r * 68 + cq + i] = x;        // zero-pad beyond VOCAB
        }
    }
    __syncthreads();
    const int tl   = tid >> 6;              // sub-tile 0..3
    const int lane = (tid >> 1) & 31;       // emulated WMMA lane
    const int jq   = (tid & 1) * 4;         // dword quad within fragment
    const int h = lane >> 4, n = lane & 15;
    uint32_t ov[4];
    #pragma unroll
    for (int j = 0; j < 4; ++j) {
        const int kl = 16 * h + 2 * (jq + j);
        ov[j] = pack_bf16(lw[kl * 68 + tl * 16 + n], lw[(kl + 1) * 68 + tl * 16 + n]);
    }
    uint4 o; o.x = ov[0]; o.y = ov[1]; o.z = ov[2]; o.w = ov[3];
    const size_t base = (((size_t)kc * 32 + (size_t)cb * 4 + tl) * 32 + lane) * 8 + jq;
    *(uint4*)(Bw + base) = o;
}

// ---------------------------------------------------------------------------
// Fused softmax(logits) @ W  +  0.5*hard-embedding mix.
// 8 wave32 / WG; WG owns 32 rows, wave owns 64 cols.  K=64 per barrier
// (16 WMMAs / barrier), DS-only split barrier, double-buffered LDS A tile,
// prefetch of next B slice into WGP$, logits streamed non-temporally.
// ---------------------------------------------------------------------------
__global__ void __launch_bounds__(256, 1) soft_token_mixer(
        const int*   __restrict__ tokens,
        const float* __restrict__ logits,
        const float* __restrict__ W,
        const uint32_t* __restrict__ Bw,
        float* __restrict__ out) {
    __shared__ alignas(16) uint32_t ldsA[2][2 * MROWS * ASTRIDE]; // [buf][64 rows x 20]
    __shared__ float ldsS[MROWS];
    __shared__ int   ldsTok[MROWS];

    const int tid  = threadIdx.x;
    const int wave = tid >> 5;
    const int lane = tid & 31;
    const int row0 = blockIdx.x * MROWS;

    if (tid < MROWS) ldsTok[tid] = tokens[row0 + tid];

    const int arow = tid >> 3;                 // staged row 0..31 (fixed per thread)
    const int q    = tid & 7;                  // K-quad within row
    const float* lrow = logits + (size_t)(row0 + arow) * VOCAB;

    v8f acc[2][4];
    #pragma unroll
    for (int mt = 0; mt < 2; ++mt)
        #pragma unroll
        for (int nt = 0; nt < 4; ++nt)
            acc[mt][nt] = (v8f)(0.0f);
    float ssum = 0.0f;

    // Stage 64 vocab entries (two K=32 sub-chunks) of exp(logit) as bf16 A tile.
    auto stage = [&](int kcs, int buf) {
        float rs = 0.0f;
        #pragma unroll
        for (int sc = 0; sc < 2; ++sc) {
            const int kb = kcs * 64 + sc * 32 + 4 * q;
            float e0, e1, e2, e3;
            if (kcs < NC64 - 1) {
                f4u x = __builtin_nontemporal_load((const f4u*)(lrow + kb));
                e0 = __expf(x[0]); e1 = __expf(x[1]); e2 = __expf(x[2]); e3 = __expf(x[3]);
            } else {                               // guarded vocab tail
                e0 = (kb + 0 < VOCAB) ? __expf(lrow[kb + 0]) : 0.0f;
                e1 = (kb + 1 < VOCAB) ? __expf(lrow[kb + 1]) : 0.0f;
                e2 = (kb + 2 < VOCAB) ? __expf(lrow[kb + 2]) : 0.0f;
                e3 = (kb + 3 < VOCAB) ? __expf(lrow[kb + 3]) : 0.0f;
            }
            rs += e0 + e1 + e2 + e3;
            uint32_t* dst = &ldsA[buf][(sc * MROWS + arow) * ASTRIDE + 2 * q];
            dst[0] = pack_bf16(e0, e1);
            dst[1] = pack_bf16(e2, e3);
        }
        rs += __shfl_xor(rs, 1);                   // row-sum across 8 staging lanes
        rs += __shfl_xor(rs, 2);
        rs += __shfl_xor(rs, 4);
        ssum += rs;
    };

    stage(0, 0);
    for (int kc = 0; kc < NC64; ++kc) {
        // B fragments: per-wave contiguous 4 KB slice per sub-chunk (L2-resident)
        FragBF bf[2][4];
        #pragma unroll
        for (int sc = 0; sc < 2; ++sc) {
            const size_t c = (size_t)(2 * kc + sc);
            const uint4* bp = (const uint4*)Bw + ((c * 32 + wave * 4) * 32 + lane) * 2;
            #pragma unroll
            for (int nt = 0; nt < 4; ++nt) {
                bf[sc][nt].u4[0] = bp[nt * 64 + 0];
                bf[sc][nt].u4[1] = bp[nt * 64 + 1];
            }
        }
        if (kc + 1 < NC64) {
            // pull next iteration's B slice L2 -> WGP$ (global_prefetch_b8)
            #pragma unroll
            for (int sc = 0; sc < 2; ++sc) {
                const size_t c = (size_t)(2 * kc + 2 + sc);
                const char* pf = (const char*)Bw + ((c * 32 + wave * 4) * 32) * 32 + lane * 128;
                __builtin_prefetch(pf, 0, 3);
            }
            stage(kc + 1, (kc + 1) & 1);           // overlap exp/pack with B loads
        }
        lds_barrier();                             // DS-only: B loads stay in flight

        FragBF af[2][2];
        const int ar = lane & 15, h4 = (lane >> 4) * 8;
        #pragma unroll
        for (int sc = 0; sc < 2; ++sc)
            #pragma unroll
            for (int mt = 0; mt < 2; ++mt) {
                const uint32_t* ap = &ldsA[kc & 1][(sc * MROWS + mt * 16 + ar) * ASTRIDE];
                af[sc][mt].u4[0] = *(const uint4*)(ap + h4);
                af[sc][mt].u4[1] = *(const uint4*)(ap + 4 + h4);
            }
        #pragma unroll
        for (int sc = 0; sc < 2; ++sc)
            #pragma unroll
            for (int mt = 0; mt < 2; ++mt)
                #pragma unroll
                for (int nt = 0; nt < 4; ++nt)
                    acc[mt][nt] = __builtin_amdgcn_wmma_f32_16x16x32_bf16(
                        false, af[sc][mt].v, false, bf[sc][nt].v,
                        (short)0, acc[mt][nt], false, false);
    }

    if (q == 0) ldsS[arow] = ssum;
    __syncthreads();

    // Epilogue: normalize, gather hard embedding, mix, streaming store.
    const int n = lane & 15, hh = lane >> 4;
    #pragma unroll
    for (int mt = 0; mt < 2; ++mt) {
        #pragma unroll
        for (int j = 0; j < 8; ++j) {
            const int row  = mt * 16 + j + 8 * hh;           // C-layout row of VGPR j
            const float sc = 0.5f / ldsS[row];               // fold SOFT_RATIO
            const int grow = row0 + row;
            const size_t hb = (size_t)ldsTok[row] * EMBED;
            #pragma unroll
            for (int nt = 0; nt < 4; ++nt) {
                const int col = wave * 64 + nt * 16 + n;
                const float v = 0.5f * W[hb + col] + acc[mt][nt][j] * sc;
                __builtin_nontemporal_store(v, out + (size_t)grow * EMBED + col);
            }
        }
    }
}

extern "C" void kernel_launch(void* const* d_in, const int* in_sizes, int n_in,
                              void* d_out, int out_size, void* d_ws, size_t ws_size,
                              hipStream_t stream) {
    const int*   tokens = (const int*)  d_in[0];   // (B,S) int32
    const float* logits = (const float*)d_in[1];   // (B,S,V) f32
    const float* W      = (const float*)d_in[2];   // (V,D) f32
    float* out = (float*)d_out;                    // (B,S,D) f32
    uint32_t* Bw = (uint32_t*)d_ws;                // needs PADC*32*32*8*4 = 51.5 MB

    prep_wswizzle<<<PADC * 8, 256, 0, stream>>>(W, Bw);
    soft_token_mixer<<<ROWS / MROWS, 256, 0, stream>>>(tokens, logits, W, Bw, out);

    (void)in_sizes; (void)n_in; (void)out_size; (void)ws_size;
}
